// MultiHeadAttention_36352603194100
// MI455X (gfx1250) — matrix operand order repaired and checked by reading, not compiled, not run on hardware
//
#include <hip/hip_runtime.h>
#include <hip/hip_bf16.h>

#define NH   16
#define LSEQ 2048
#define DM   1024
#define DH   64
#define NB   2

typedef _Float16 f16;
typedef __attribute__((ext_vector_type(16))) _Float16 v16h;
typedef __attribute__((ext_vector_type(8)))  _Float16 v8h;
typedef __attribute__((ext_vector_type(8)))  float    v8f;
typedef __attribute__((ext_vector_type(4)))  float    f4;
typedef __attribute__((ext_vector_type(4)))  unsigned int u32x4;
typedef __attribute__((ext_vector_type(4)))  int      i32x4;

#if defined(__has_builtin)
#if __has_builtin(__builtin_amdgcn_global_load_async_to_lds_b128) && \
    __has_builtin(__builtin_amdgcn_s_wait_asynccnt)
#define ATTN_ASYNC_MASK 1
#endif
#endif

#ifdef ATTN_ASYNC_MASK
typedef __attribute__((address_space(1))) i32x4 as1_i32x4;  // global (__device__) int4
typedef __attribute__((address_space(3))) i32x4 as3_i32x4;  // LDS (__shared__) int4
#endif

union V16U { v16h v; v8h h[2]; };

static __device__ __forceinline__ int lane_id() { return (int)(threadIdx.x & 31u); }

static __device__ __forceinline__ v8f wmma16(v16h a, v16h b, v8f c) {
  // D = A(16x32 f16) * B(32x16 f16) + C(16x16 f32)
  return __builtin_amdgcn_wmma_f32_16x16x32_f16(false, a, false, b, (short)0, c, false, false);
}

// ---- A-matrix fragment (M=16, K=32). Source row-major [rows x ld] f16.
// Lane layout (ISA 7.12.2): m = lane%16; v[0..7] = K[k0+half*8 ..], v[8..15] = K[k0+16+half*8 ..]
static __device__ __forceinline__ v16h afrag_f16(const f16* __restrict__ base, int ld, int rowbase, int k0) {
  int lane = lane_id();
  int half = lane >> 4;
  const f16* p = base + (size_t)(rowbase + (lane & 15)) * ld + k0 + half * 8;
  V16U u;
  u.h[0] = *(const v8h*)p;
  u.h[1] = *(const v8h*)(p + 16);
  return u.v;
}

static __device__ __forceinline__ v16h afrag_f32(const float* __restrict__ base, int ld, int rowbase, int k0) {
  int lane = lane_id();
  int half = lane >> 4;
  const float* p = base + (size_t)(rowbase + (lane & 15)) * ld + k0 + half * 8;
  f4 x0 = *(const f4*)p;
  f4 x1 = *(const f4*)(p + 4);
  f4 x2 = *(const f4*)(p + 16);
  f4 x3 = *(const f4*)(p + 20);
  v16h v;
#pragma unroll
  for (int i = 0; i < 4; ++i) {
    v[i]      = (f16)x0[i];
    v[4 + i]  = (f16)x1[i];
    v[8 + i]  = (f16)x2[i];
    v[12 + i] = (f16)x3[i];
  }
  return v;
}

// ---- B-matrix fragment (K=32, N=16). Source is B^T row-major: bt[n][k], k contiguous.
// Lane layout: n = lane%16; v[0..7] = B[k0+half*8 .., n], v[8..15] = B[k0+16+half*8 .., n]
static __device__ __forceinline__ v16h bfrag_f16(const f16* __restrict__ bt, int ld, int nbase, int k0) {
  int lane = lane_id();
  int half = lane >> 4;
  const f16* p = bt + (size_t)(nbase + (lane & 15)) * ld + k0 + half * 8;
  V16U u;
  u.h[0] = *(const v8h*)p;
  u.h[1] = *(const v8h*)(p + 16);
  return u.v;
}

static __device__ __forceinline__ float rmax16(float v) {
#pragma unroll
  for (int m = 1; m < 16; m <<= 1) v = fmaxf(v, __shfl_xor(v, m, 16));
  return v;
}
static __device__ __forceinline__ float rsum16(float v) {
#pragma unroll
  for (int m = 1; m < 16; m <<= 1) v += __shfl_xor(v, m, 16);
  return v;
}

// ======================= weight transpose f32[K,N] -> f16[N,K] ==================
__global__ __launch_bounds__(256) void transpose_w_kernel(const float* __restrict__ src,
                                                          f16* __restrict__ dst) {
  int k = blockIdx.x * 16 + threadIdx.x;
  int n = blockIdx.y * 16 + threadIdx.y;
  dst[(size_t)n * DM + k] = (f16)src[(size_t)k * DM + n];
}

// ======================= QKV projection: f32 A @ f16 W^T + bias =================
// mode 0: dst[(b*NH+h)*LSEQ*DH + l*DH + d]  (q,k layout [B,H,L,64])
// mode 1: dst[(b*NH+h)*DH*LSEQ + d*LSEQ + l] (v transposed [B,H,64,L])
static __device__ __forceinline__ void store_qkv_tile(v8f c, int mrow, int ncol,
                                                      const float* __restrict__ bias,
                                                      f16* __restrict__ dst, int mode) {
  int lane = lane_id();
  int half = lane >> 4;
  int nc = ncol + (lane & 15);
  float bv = bias[nc];
  int h = nc >> 6, d = nc & 63;
#pragma unroll
  for (int j = 0; j < 8; ++j) {
    int m = mrow + j + 8 * half;          // global row = b*LSEQ + l
    int b = m >> 11, l = m & (LSEQ - 1);
    float val = c[j] + bv;
    if (mode == 0)
      dst[((size_t)(b * NH + h) * LSEQ + l) * DH + d] = (f16)val;
    else
      dst[((size_t)(b * NH + h) * DH + d) * LSEQ + l] = (f16)val;
  }
}

__global__ __launch_bounds__(256) void proj_kernel(const float* __restrict__ A,
                                                   const f16* __restrict__ Wt,
                                                   const float* __restrict__ bias,
                                                   f16* __restrict__ dst, int mode) {
  int w = (int)(threadIdx.x >> 5);
  int mbase = blockIdx.x * 128 + (w & 3) * 32;
  int nbase = blockIdx.y * 64 + (w >> 2) * 32;
  v8f c00 = {}, c01 = {}, c10 = {}, c11 = {};
  for (int k0 = 0; k0 < DM; k0 += 32) {
    v16h a0 = afrag_f32(A, DM, mbase, k0);
    v16h a1 = afrag_f32(A, DM, mbase + 16, k0);
    v16h b0 = bfrag_f16(Wt, DM, nbase, k0);
    v16h b1 = bfrag_f16(Wt, DM, nbase + 16, k0);
    c00 = wmma16(a0, b0, c00);
    c01 = wmma16(a0, b1, c01);
    c10 = wmma16(a1, b0, c10);
    c11 = wmma16(a1, b1, c11);
  }
  store_qkv_tile(c00, mbase, nbase, bias, dst, mode);
  store_qkv_tile(c01, mbase, nbase + 16, bias, dst, mode);
  store_qkv_tile(c10, mbase + 16, nbase, bias, dst, mode);
  store_qkv_tile(c11, mbase + 16, nbase + 16, bias, dst, mode);
}

// ======================= attention (flash-style) ================================
// One block = one (b, 16-query tile); 16 waves = 16 heads. The mask's innermost
// dim is h (16 contiguous bytes), so the block cooperatively stages the full
// [16q x 32key x 16h] mask tile (double-buffered, async-to-LDS when available).
// q,k: [B,H,L,64] f16 ; vt: [B,H,64,L] f16 ; mask: [B,L,L,H] u8 ; ctx: [B*L, 1024] f16
__global__ __launch_bounds__(512) void attn_kernel(const f16* __restrict__ q,
                                                   const f16* __restrict__ k,
                                                   const f16* __restrict__ vt,
                                                   const unsigned char* __restrict__ mask,
                                                   f16* __restrict__ ctx) {
  __shared__ __align__(16) unsigned char mt[2][16][32][NH];  // 16 KB double-buffered mask tile
  __shared__ f16 ptile[NH][16][32];                          // 16 KB per-wave P tiles

  int tid = (int)threadIdx.x;
  int w = tid >> 5;            // wave id == head id
  int lane = lane_id();
  int half = lane >> 4;
  int h = w;
  int b = (int)blockIdx.y;
  int q0 = (int)blockIdx.x * 16;

  const f16* qh = q + (size_t)(b * NH + h) * LSEQ * DH;
  const f16* kh = k + (size_t)(b * NH + h) * LSEQ * DH;
  const f16* vh = vt + (size_t)(b * NH + h) * DH * LSEQ;
  const unsigned char* mb = mask + (size_t)b * LSEQ * LSEQ * NH;

  // cooperative mask-tile coordinates: 512 threads x 16B = 16q x 32key x 16h
  int mr = tid >> 5;           // query row 0..15
  int mc = tid & 31;           // key col 0..31
  const unsigned char* mrow_base = mb + ((size_t)(q0 + mr) * LSEQ + mc) * NH;

  // q fragments: d=[0,32) and [32,64), kept resident
  v16h qa0 = afrag_f16(qh, DH, q0, 0);
  v16h qa1 = afrag_f16(qh, DH, q0, 32);

  v8f o[4] = {v8f{}, v8f{}, v8f{}, v8f{}};
  float mrun[8], lrun[8];
#pragma unroll
  for (int j = 0; j < 8; ++j) { mrun[j] = -1e30f; lrun[j] = 0.0f; }

  // stage mask tile for chunk 0 into buffer 0
#ifdef ATTN_ASYNC_MASK
  __builtin_amdgcn_global_load_async_to_lds_b128(
      (as1_i32x4*)(mrow_base), (as3_i32x4*)&mt[0][mr][mc][0], 0, 0);
#else
  *(u32x4*)&mt[0][mr][mc][0] = *(const u32x4*)(mrow_base);
#endif

  for (int kk0 = 0; kk0 < LSEQ; kk0 += 32) {
    int buf = (kk0 >> 5) & 1;
    // stage next chunk's mask tile (last iteration: harmless dummy refetch of chunk 0
    // so exactly one async op stays in flight and the wait count is a constant)
    int nk = (kk0 + 32 < LSEQ) ? (kk0 + 32) : 0;
#ifdef ATTN_ASYNC_MASK
    __builtin_amdgcn_global_load_async_to_lds_b128(
        (as1_i32x4*)(mrow_base + (size_t)nk * NH), (as3_i32x4*)&mt[buf ^ 1][mr][mc][0], 0, 0);
#else
    *(u32x4*)&mt[buf ^ 1][mr][mc][0] = *(const u32x4*)(mrow_base + (size_t)nk * NH);
#endif

    // ---- scores for 32 keys: two 16x16 tiles, K-dim = d (64 = 2x32)
    v8f s0 = {}, s1 = {};
    s0 = wmma16(qa0, bfrag_f16(kh, DH, kk0, 0), s0);
    s0 = wmma16(qa1, bfrag_f16(kh, DH, kk0, 32), s0);
    s1 = wmma16(qa0, bfrag_f16(kh, DH, kk0 + 16, 0), s1);
    s1 = wmma16(qa1, bfrag_f16(kh, DH, kk0 + 16, 32), s1);

#ifdef ATTN_ASYNC_MASK
    __builtin_amdgcn_s_wait_asynccnt(1);  // in-order: current chunk's tile has landed
#endif
    __syncthreads();   // mask tile visible to all waves

    // ---- mask + online softmax (rows j+8*half, cols stripe across 16 lanes)
    float oscale[8];
#pragma unroll
    for (int j = 0; j < 8; ++j) {
      int r = j + 8 * half;
      unsigned char msk0 = mt[buf][r][lane & 15][h];
      unsigned char msk1 = mt[buf][r][(lane & 15) + 16][h];
      float x0 = msk0 ? 1e-9f : s0[j];
      float x1 = msk1 ? 1e-9f : s1[j];
      float cm = rmax16(fmaxf(x0, x1));
      float mn = fmaxf(mrun[j], cm);
      float corr = __expf(mrun[j] - mn);
      float p0 = __expf(x0 - mn);
      float p1 = __expf(x1 - mn);
      lrun[j] = lrun[j] * corr + rsum16(p0 + p1);
      mrun[j] = mn;
      oscale[j] = corr;
      ptile[w][r][lane & 15] = (f16)p0;
      ptile[w][r][(lane & 15) + 16] = (f16)p1;
    }
#pragma unroll
    for (int t = 0; t < 4; ++t)
#pragma unroll
      for (int j = 0; j < 8; ++j) o[t][j] *= oscale[j];

    __syncthreads();   // P tile stored + all mask reads done before next overwrite
    // P as A-fragment (16 queries x 32 keys) from LDS
    v16h pa = afrag_f16(&ptile[w][0][0], 32, 0, 0);

    // ---- O += P @ V : B = v^T rows d, contiguous along keys
#pragma unroll
    for (int t = 0; t < 4; ++t)
      o[t] = wmma16(pa, bfrag_f16(vh, LSEQ, t * 16, kk0), o[t]);
  }

  // ---- normalize and store ctx [B*L, NH*64] f16
  float inv[8];
#pragma unroll
  for (int j = 0; j < 8; ++j) inv[j] = 1.0f / lrun[j];
#pragma unroll
  for (int t = 0; t < 4; ++t) {
    int n = h * DH + t * 16 + (lane & 15);
#pragma unroll
    for (int j = 0; j < 8; ++j) {
      int qrow = q0 + j + 8 * half;
      ctx[(size_t)(b * LSEQ + qrow) * DM + n] = (f16)(o[t][j] * inv[j]);
    }
  }
}

// ======================= output projection: f16 ctx @ f16 Wo^T + bo -> f32 ======
__global__ __launch_bounds__(256) void outproj_kernel(const f16* __restrict__ A,
                                                      const f16* __restrict__ Wt,
                                                      const float* __restrict__ bias,
                                                      float* __restrict__ out) {
  int w = (int)(threadIdx.x >> 5);
  int lane = lane_id();
  int half = lane >> 4;
  int mbase = blockIdx.x * 128 + (w & 3) * 32;
  int nbase = blockIdx.y * 64 + (w >> 2) * 32;
  v8f c00 = {}, c01 = {}, c10 = {}, c11 = {};
  for (int k0 = 0; k0 < DM; k0 += 32) {
    v16h a0 = afrag_f16(A, DM, mbase, k0);
    v16h a1 = afrag_f16(A, DM, mbase + 16, k0);
    v16h b0 = bfrag_f16(Wt, DM, nbase, k0);
    v16h b1 = bfrag_f16(Wt, DM, nbase + 16, k0);
    c00 = wmma16(a0, b0, c00);
    c01 = wmma16(a0, b1, c01);
    c10 = wmma16(a1, b0, c10);
    c11 = wmma16(a1, b1, c11);
  }
  v8f cc[4] = {c00, c01, c10, c11};
#pragma unroll
  for (int t = 0; t < 4; ++t) {
    int mrow = mbase + (t >> 1) * 16;
    int nc = nbase + (t & 1) * 16 + (lane & 15);
    float bv = bias[nc];
#pragma unroll
    for (int j = 0; j < 8; ++j) {
      int m = mrow + j + 8 * half;
      out[(size_t)m * DM + nc] = cc[t][j] + bv;
    }
  }
}

extern "C" void kernel_launch(void* const* d_in, const int* in_sizes, int n_in,
                              void* d_out, int out_size, void* d_ws, size_t ws_size,
                              hipStream_t stream) {
  (void)in_sizes; (void)n_in; (void)out_size; (void)ws_size;
  const float* Q  = (const float*)d_in[0];
  const float* K  = (const float*)d_in[1];
  const float* V  = (const float*)d_in[2];
  const unsigned char* mask = (const unsigned char*)d_in[3];
  const float* WQ = (const float*)d_in[4];
  const float* bQ = (const float*)d_in[5];
  const float* WK = (const float*)d_in[6];
  const float* bK = (const float*)d_in[7];
  const float* WV = (const float*)d_in[8];
  const float* bV = (const float*)d_in[9];
  const float* Wo = (const float*)d_in[10];
  const float* bo = (const float*)d_in[11];
  float* out = (float*)d_out;

  char* ws = (char*)d_ws;
  f16* WQt = (f16*)(ws + ((size_t)0 << 20));   // 2 MB each
  f16* WKt = (f16*)(ws + ((size_t)2 << 20));
  f16* WVt = (f16*)(ws + ((size_t)4 << 20));
  f16* WOt = (f16*)(ws + ((size_t)6 << 20));
  f16* qh  = (f16*)(ws + ((size_t)8 << 20));   // 8 MB each
  f16* kh  = (f16*)(ws + ((size_t)16 << 20));
  f16* vt  = (f16*)(ws + ((size_t)24 << 20));
  f16* ctx = (f16*)(ws + ((size_t)32 << 20));  // 8 MB; total 40 MB

  dim3 tb(16, 16), tg(DM / 16, DM / 16);
  transpose_w_kernel<<<tg, tb, 0, stream>>>(WQ, WQt);
  transpose_w_kernel<<<tg, tb, 0, stream>>>(WK, WKt);
  transpose_w_kernel<<<tg, tb, 0, stream>>>(WV, WVt);
  transpose_w_kernel<<<tg, tb, 0, stream>>>(Wo, WOt);

  dim3 gg(NB * LSEQ / 128, DM / 64);  // 32 x 16
  proj_kernel<<<gg, 256, 0, stream>>>(Q, WQt, bQ, qh, 0);
  proj_kernel<<<gg, 256, 0, stream>>>(K, WKt, bK, kh, 0);
  proj_kernel<<<gg, 256, 0, stream>>>(V, WVt, bV, vt, 1);

  dim3 ag(LSEQ / 16, NB);  // 128 x 2 blocks; 16 waves/block = 16 heads
  attn_kernel<<<ag, 512, 0, stream>>>(qh, kh, vt, mask, ctx);

  outproj_kernel<<<gg, 256, 0, stream>>>(ctx, WOt, bo, out);
}
